// W_MSA_87514253623951
// MI455X (gfx1250) — compile-verified, matrix operand order checked by reading, not run on hardware
//
#include <hip/hip_runtime.h>
#include <math.h>

// ---------------------------------------------------------------------------
// Problem constants (Swin window-MSA block)
// ---------------------------------------------------------------------------
#define Bsz   8
#define Pdim  56
#define WSz   7
#define Cdim  1024
#define Hn    16
#define Dh    64
#define Ttok  (Pdim * Pdim)                  // 3136
#define NWin  ((Pdim / WSz) * (Pdim / WSz))  // 64 windows / image
#define NWIN  (Bsz * NWin)                   // 512 windows total
#define WT    (WSz * WSz)                    // 49 tokens / window
#define MT    (NWIN * WT)                    // 25088 rows (window order) == B*T
#define ATT_SCALE 0.03125f                   // C^-0.5 = 1024^-0.5
#define LN_EPS 1e-5f

typedef __bf16 v16bf __attribute__((ext_vector_type(16)));
typedef __bf16 bf16x2 __attribute__((ext_vector_type(2)));
typedef float  v8f   __attribute__((ext_vector_type(8)));

__device__ __forceinline__ __bf16 f2bf(float f) { return (__bf16)f; }

// Async global->LDS staging path (gfx1250 GLOBAL_LOAD_ASYNC_TO_LDS_B128),
// guarded so the file compiles on toolchains without the builtins.
// Probe-derived signature: (int4 AS1* gsrc, int4 AS3* ldst, imm offset, imm cpol)
#if defined(__AMDGCN__) &&                                        \
    __has_builtin(__builtin_amdgcn_global_load_async_to_lds_b128) && \
    __has_builtin(__builtin_amdgcn_s_wait_asynccnt)
#define HAVE_ASYNC_LDS 1
typedef int i4v __attribute__((ext_vector_type(4)));
typedef __attribute__((address_space(1))) i4v* gvec_ptr;
typedef __attribute__((address_space(3))) i4v* lvec_ptr;
#else
#define HAVE_ASYNC_LDS 0
#endif

// ---------------------------------------------------------------------------
// WMMA fragment loader per CDNA5 ISA 7.12.2 (16-bit A 16x32 layout).
// k-major source: base[row][k], row stride = ldk elements, k pairs contiguous.
// Used for A operands (row = M) and for B operands staged n-major (row = N).
// ---------------------------------------------------------------------------
__device__ __forceinline__ v16bf load_frag_kmajor(const __bf16* base, int ldk,
                                                  int row0, int k0, int lane) {
  const int m = row0 + (lane & 15);
  const int half = (lane >> 4) & 1;
  const __bf16* p = base + m * ldk + k0 + (half << 3);
  v16bf f;
#pragma unroll
  for (int v = 0; v < 8; ++v) {
    const int k = ((v >> 2) << 4) + ((v & 3) << 1);  // 0,2,4,6,16,18,20,22
    bf16x2 t = *(const bf16x2*)(p + k);
    f[2 * v]     = t.x;
    f[2 * v + 1] = t.y;
  }
  return f;
}

__device__ __forceinline__ v8f wmma_bf16(v16bf a, v16bf b, v8f c) {
  return __builtin_amdgcn_wmma_f32_16x16x32_bf16(false, a, false, b,
                                                 (short)0, c, false, false);
}

// ---------------------------------------------------------------------------
// Weight packing: fp32 -> bf16, stored TRANSPOSED n-major: Bt[n][k].
// mode 0: src = [H, C, D] per-head QKV weights; n = h*64 + d, k = c.
// mode 1: src = [C_in, C_out] row-major;        n = c_out,    k = c_in.
// ---------------------------------------------------------------------------
__global__ __launch_bounds__(256) void pack_kernel(const float* __restrict__ src,
                                                   __bf16* __restrict__ dst,
                                                   int mode) {
  const int i = blockIdx.x * 256 + threadIdx.x;  // output index over C*C
  const int n = i >> 10, k = i & 1023;
  float v;
  if (mode == 0) {
    const int h = n >> 6, d = n & 63;
    v = src[((size_t)h * Cdim + k) * Dh + d];
  } else {
    v = src[(size_t)k * Cdim + n];
  }
  dst[i] = f2bf(v);
}

// ---------------------------------------------------------------------------
// Block reduction helper (256 threads)
// ---------------------------------------------------------------------------
__device__ __forceinline__ float block_sum256(float v, float* red) {
  const int tid = threadIdx.x;
  red[tid] = v;
  __syncthreads();
#pragma unroll
  for (int s = 128; s > 0; s >>= 1) {
    if (tid < s) red[tid] += red[tid + s];
    __syncthreads();
  }
  const float r = red[0];
  __syncthreads();
  return r;
}

// ---------------------------------------------------------------------------
// LN1 + window partition gather; writes fp32 residual copy, bf16 activations,
// and the window-row -> token rowmap used later for the merge scatter.
// ---------------------------------------------------------------------------
__global__ __launch_bounds__(256) void ln1_kernel(const float* __restrict__ x,
                                                  const float* __restrict__ g,
                                                  const float* __restrict__ b,
                                                  float* __restrict__ wout,
                                                  __bf16* __restrict__ wbf,
                                                  int* __restrict__ rowmap) {
  __shared__ float red[256];
  const int r = blockIdx.x;                 // window-order row 0..MT-1
  const int win = r / WT, t = r - win * WT;
  const int bimg = win >> 6, wloc = win & 63;
  const int wy = wloc >> 3, wx = wloc & 7;
  const int iy = t / WSz, ix = t - iy * WSz;
  const int tok = (wy * WSz + iy) * Pdim + (wx * WSz + ix);
  const size_t src = ((size_t)bimg * Ttok + tok) * Cdim;
  if (threadIdx.x == 0) rowmap[r] = bimg * Ttok + tok;

  float lx[4];
  float s = 0.f;
#pragma unroll
  for (int i = 0; i < 4; ++i) {
    lx[i] = x[src + threadIdx.x + i * 256];
    s += lx[i];
  }
  const float mean = block_sum256(s, red) * (1.0f / Cdim);
  float v = 0.f;
#pragma unroll
  for (int i = 0; i < 4; ++i) {
    const float d = lx[i] - mean;
    v += d * d;
  }
  const float rstd = rsqrtf(block_sum256(v, red) * (1.0f / Cdim) + LN_EPS);
  const size_t dst = (size_t)r * Cdim;
#pragma unroll
  for (int i = 0; i < 4; ++i) {
    const int c = threadIdx.x + i * 256;
    const float y = (lx[i] - mean) * rstd * g[c] + b[c];
    wout[dst + c] = y;
    wbf[dst + c] = f2bf(y);
  }
}

// LN2 over token-order m; bf16 output only.
__global__ __launch_bounds__(256) void ln2_kernel(const float* __restrict__ m,
                                                  const float* __restrict__ g,
                                                  const float* __restrict__ b,
                                                  __bf16* __restrict__ out) {
  __shared__ float red[256];
  const size_t src = (size_t)blockIdx.x * Cdim;
  float lx[4];
  float s = 0.f;
#pragma unroll
  for (int i = 0; i < 4; ++i) {
    lx[i] = m[src + threadIdx.x + i * 256];
    s += lx[i];
  }
  const float mean = block_sum256(s, red) * (1.0f / Cdim);
  float v = 0.f;
#pragma unroll
  for (int i = 0; i < 4; ++i) {
    const float d = lx[i] - mean;
    v += d * d;
  }
  const float rstd = rsqrtf(block_sum256(v, red) * (1.0f / Cdim) + LN_EPS);
#pragma unroll
  for (int i = 0; i < 4; ++i) {
    const int c = threadIdx.x + i * 256;
    out[src + c] = f2bf((lx[i] - mean) * rstd * g[c] + b[c]);
  }
}

// ---------------------------------------------------------------------------
// Generic WMMA GEMM: C[M,N] = A[M,K] * Bt[N,K]^T, bf16 in, fp32 accumulate.
// Block = 256 threads (8 waves), tile 128x128x32, wave grid 2x4,
// each wave owns a 64x32 sub-tile = 4x2 accumulator fragments.
// Double-buffered LDS staging; async global->LDS copies when available.
// ---------------------------------------------------------------------------
#define EPI_BF16          0  // out_bf = acc
#define EPI_GELU_BF16     1  // out_bf = gelu(acc + bias)
#define EPI_BIAS_RES_SCAT 2  // out_f[rowmap[m]] = resid[m] + acc + bias (merge)
#define EPI_BIAS_RES      3  // out_f[m] = resid[m] + acc + bias

template <int EPI>
__global__ __launch_bounds__(256) void gemm_wmma(
    const __bf16* __restrict__ A, const __bf16* __restrict__ Bt,
    float* __restrict__ outf, __bf16* __restrict__ outb,
    const float* __restrict__ bias, const float* __restrict__ resid,
    const int* __restrict__ rowmap, int M, int N, int K) {
  constexpr int BM = 128, BN = 128, BK = 32, LDK = BK + 8;  // 80B rows
  __shared__ __bf16 As[2][BM * LDK];
  __shared__ __bf16 Bs[2][BN * LDK];  // n-major rows (Bt layout), k contiguous

  const int tid = threadIdx.x;
  const int lane = tid & 31;
  const int wave = tid >> 5;
  const int wm = wave >> 2;  // 0..1
  const int wn = wave & 3;   // 0..3
  const int m0 = blockIdx.y * BM;
  const int n0 = blockIdx.x * BN;

  v8f acc[4][2];
#pragma unroll
  for (int i = 0; i < 4; ++i)
#pragma unroll
    for (int j = 0; j < 2; ++j) acc[i][j] = (v8f){0, 0, 0, 0, 0, 0, 0, 0};

  const int row = tid >> 1, seg = (tid & 1) << 4;  // 128 rows x 2 segs of 16

  auto stage = [&](int buf, int k0) {
    const __bf16* gA = A + (size_t)(m0 + row) * K + k0 + seg;
    const __bf16* gB = Bt + (size_t)(n0 + row) * K + k0 + seg;
    __bf16* lA = &As[buf][row * LDK + seg];
    __bf16* lB = &Bs[buf][row * LDK + seg];
#if HAVE_ASYNC_LDS
    __builtin_amdgcn_global_load_async_to_lds_b128((gvec_ptr)(void*)gA,
                                                   (lvec_ptr)(void*)lA, 0, 0);
    __builtin_amdgcn_global_load_async_to_lds_b128((gvec_ptr)(void*)gA,
                                                   (lvec_ptr)(void*)lA, 16, 0);
    __builtin_amdgcn_global_load_async_to_lds_b128((gvec_ptr)(void*)gB,
                                                   (lvec_ptr)(void*)lB, 0, 0);
    __builtin_amdgcn_global_load_async_to_lds_b128((gvec_ptr)(void*)gB,
                                                   (lvec_ptr)(void*)lB, 16, 0);
#else
    *(uint4*)lA       = *(const uint4*)gA;
    *(uint4*)(lA + 8) = *(const uint4*)(gA + 8);
    *(uint4*)lB       = *(const uint4*)gB;
    *(uint4*)(lB + 8) = *(const uint4*)(gB + 8);
#endif
  };

  stage(0, 0);
#if HAVE_ASYNC_LDS
  __builtin_amdgcn_s_wait_asynccnt(0);
#endif
  __syncthreads();

  int cur = 0;
  for (int k0 = 0; k0 < K; k0 += BK) {
    if (k0 + BK < K) stage(cur ^ 1, k0 + BK);

    v16bf af[4], bfv[2];
#pragma unroll
    for (int fm = 0; fm < 4; ++fm)
      af[fm] = load_frag_kmajor(As[cur], LDK, wm * 64 + fm * 16, 0, lane);
#pragma unroll
    for (int fn = 0; fn < 2; ++fn)
      bfv[fn] = load_frag_kmajor(Bs[cur], LDK, wn * 32 + fn * 16, 0, lane);
#pragma unroll
    for (int fm = 0; fm < 4; ++fm)
#pragma unroll
      for (int fn = 0; fn < 2; ++fn)
        acc[fm][fn] = wmma_bf16(af[fm], bfv[fn], acc[fm][fn]);

#if HAVE_ASYNC_LDS
    __builtin_amdgcn_s_wait_asynccnt(0);
#endif
    __syncthreads();
    cur ^= 1;
  }

  // ---- epilogue (C layout: VGPR r -> row r (lanes 0-15) / r+8 (lanes 16-31))
  const int half8 = ((lane >> 4) & 1) << 3;
  const int cl = lane & 15;
#pragma unroll
  for (int fm = 0; fm < 4; ++fm) {
#pragma unroll
    for (int fn = 0; fn < 2; ++fn) {
#pragma unroll
      for (int r = 0; r < 8; ++r) {
        const int rr = m0 + wm * 64 + fm * 16 + r + half8;
        const int cc = n0 + wn * 32 + fn * 16 + cl;
        const float a = acc[fm][fn][r];
        if (EPI == EPI_BF16) {
          outb[(size_t)rr * N + cc] = f2bf(a);
        } else if (EPI == EPI_GELU_BF16) {
          const float xv = a + bias[cc];
          outb[(size_t)rr * N + cc] =
              f2bf(0.5f * xv * (1.0f + erff(xv * 0.7071067811865476f)));
        } else if (EPI == EPI_BIAS_RES_SCAT) {
          const float v = resid[(size_t)rr * N + cc] + a + bias[cc];
          outf[(size_t)rowmap[rr] * N + cc] = v;
        } else {  // EPI_BIAS_RES
          outf[(size_t)rr * N + cc] = resid[(size_t)rr * N + cc] + a + bias[cc];
        }
      }
    }
  }
}

// ---------------------------------------------------------------------------
// Attention per (window, head): 49 tokens padded to 64.
// S = Q*K^T * scale -> masked softmax -> O = P*V, all via WMMA.
// V is staged transposed (Vt[d][t]) so both matmuls use the vectorized
// k-major fragment loader. Block = 128 threads (4 waves).
// ---------------------------------------------------------------------------
__global__ __launch_bounds__(128) void attn_kernel(const __bf16* __restrict__ qb,
                                                   const __bf16* __restrict__ kb,
                                                   const __bf16* __restrict__ vb,
                                                   __bf16* __restrict__ ob) {
  constexpr int LD = 72;  // 144B rows, 16B friendly, conflict-spread
  __shared__ __bf16 Qs[64 * LD];
  __shared__ __bf16 Ks[64 * LD];  // later reused to hold P (bf16)
  __shared__ __bf16 Vt[64 * LD];  // transposed: Vt[d][t]
  __shared__ float Ss[64 * 66];

  const int win = blockIdx.x >> 4;
  const int h = blockIdx.x & 15;
  const size_t base = ((size_t)win * WT) * Cdim + h * Dh;
  const int tid = threadIdx.x, lane = tid & 31, wave = tid >> 5;

  for (int idx = tid; idx < 64 * 64; idx += 128) {
    const int t = idx >> 6, d = idx & 63;
    __bf16 qv = (__bf16)0.f, kv = (__bf16)0.f, vv = (__bf16)0.f;
    if (t < WT) {
      const size_t g = base + (size_t)t * Cdim + d;
      qv = qb[g]; kv = kb[g]; vv = vb[g];
    }
    Qs[t * LD + d] = qv;
    Ks[t * LD + d] = kv;
    Vt[d * LD + t] = vv;
  }
  __syncthreads();

  // S = Q * K^T (K rows are n-major for the B operand), scale + column mask
  const int m0 = wave * 16;
  const int half8 = ((lane >> 4) & 1) << 3;
  const int cl = lane & 15;
#pragma unroll
  for (int fn = 0; fn < 4; ++fn) {
    v8f c = (v8f){0, 0, 0, 0, 0, 0, 0, 0};
#pragma unroll
    for (int kk = 0; kk < 64; kk += 32) {
      v16bf a = load_frag_kmajor(Qs, LD, m0, kk, lane);
      v16bf b = load_frag_kmajor(Ks, LD, fn * 16, kk, lane);
      c = wmma_bf16(a, b, c);
    }
    const int col = fn * 16 + cl;
#pragma unroll
    for (int r = 0; r < 8; ++r) {
      const int rowi = m0 + r + half8;
      Ss[rowi * 66 + col] = (col < WT) ? c[r] * ATT_SCALE : -1e30f;
    }
  }
  __syncthreads();

  // softmax rows -> P (bf16) overwriting Ks
  if (tid < 64) {
    float mx = -3.0e38f;
    for (int j = 0; j < 64; ++j) mx = fmaxf(mx, Ss[tid * 66 + j]);
    float sum = 0.f;
    float e[64];
    for (int j = 0; j < 64; ++j) {
      e[j] = __expf(Ss[tid * 66 + j] - mx);
      sum += e[j];
    }
    const float inv = 1.0f / sum;
    for (int j = 0; j < 64; ++j) Ks[tid * LD + j] = f2bf(e[j] * inv);
  }
  __syncthreads();

  // O = P * V: B[k][n] = V[k][n] = Vt[n][k] -> k-major loader on Vt rows
#pragma unroll
  for (int fn = 0; fn < 4; ++fn) {
    v8f c = (v8f){0, 0, 0, 0, 0, 0, 0, 0};
#pragma unroll
    for (int kk = 0; kk < 64; kk += 32) {
      v16bf a = load_frag_kmajor(Ks, LD, m0, kk, lane);
      v16bf b = load_frag_kmajor(Vt, LD, fn * 16, kk, lane);
      c = wmma_bf16(a, b, c);
    }
    const int col = fn * 16 + cl;
#pragma unroll
    for (int r = 0; r < 8; ++r) {
      const int rowi = m0 + r + half8;
      if (rowi < WT) ob[base + (size_t)rowi * Cdim + col] = f2bf(c[r]);
    }
  }
}

// ---------------------------------------------------------------------------
// Host orchestration
// ---------------------------------------------------------------------------
extern "C" void kernel_launch(void* const* d_in, const int* in_sizes, int n_in,
                              void* d_out, int out_size, void* d_ws, size_t ws_size,
                              hipStream_t stream) {
  const float* x  = (const float*)d_in[0];
  const float* g1 = (const float*)d_in[1];
  const float* b1 = (const float*)d_in[2];
  const float* Wq = (const float*)d_in[3];
  const float* Wk = (const float*)d_in[4];
  const float* Wv = (const float*)d_in[5];
  const float* Wo = (const float*)d_in[6];
  const float* bo = (const float*)d_in[7];
  const float* g2 = (const float*)d_in[8];
  const float* b2 = (const float*)d_in[9];
  const float* W1 = (const float*)d_in[10];
  const float* bf1 = (const float*)d_in[11];
  const float* W2 = (const float*)d_in[12];
  const float* bf2 = (const float*)d_in[13];
  float* out = (float*)d_out;

  const size_t MTC = (size_t)MT * Cdim;      // 25,690,112 elements
  const size_t F32B = MTC * 4, BFB = MTC * 2;
  const size_t WB = (size_t)Cdim * Cdim * 2; // packed weight bytes
  auto al = [](size_t v) { return (v + 255) & ~(size_t)255; };

  char* ws = (char*)d_ws;
  size_t off = 0;
  float*  wwin_f32 = (float*)(ws + off);  off = al(off + F32B);
  __bf16* wwin_bf  = (__bf16*)(ws + off); off = al(off + BFB);  // reused as LN2 out
  __bf16* qb = (__bf16*)(ws + off);       off = al(off + BFB);  // reused as fc1 out
  __bf16* kb = (__bf16*)(ws + off);       off = al(off + BFB);
  __bf16* vb = (__bf16*)(ws + off);       off = al(off + BFB);
  __bf16* obuf = (__bf16*)(ws + off);     off = al(off + BFB);
  float*  m_f32 = (float*)(ws + off);     off = al(off + F32B);
  __bf16* Bq = (__bf16*)(ws + off);       off = al(off + WB);
  __bf16* Bk = (__bf16*)(ws + off);       off = al(off + WB);
  __bf16* Bv = (__bf16*)(ws + off);       off = al(off + WB);
  __bf16* Bo = (__bf16*)(ws + off);       off = al(off + WB);
  __bf16* B1 = (__bf16*)(ws + off);       off = al(off + WB);
  __bf16* B2 = (__bf16*)(ws + off);       off = al(off + WB);
  int*    rowmap = (int*)(ws + off);      off = al(off + (size_t)MT * 4);
  __bf16* mln = wwin_bf;  // reuse
  __bf16* h1  = qb;       // reuse

  const dim3 packG(Cdim * Cdim / 256), packB(256);
  pack_kernel<<<packG, packB, 0, stream>>>(Wq, Bq, 0);
  pack_kernel<<<packG, packB, 0, stream>>>(Wk, Bk, 0);
  pack_kernel<<<packG, packB, 0, stream>>>(Wv, Bv, 0);
  pack_kernel<<<packG, packB, 0, stream>>>(Wo, Bo, 1);
  pack_kernel<<<packG, packB, 0, stream>>>(W1, B1, 1);
  pack_kernel<<<packG, packB, 0, stream>>>(W2, B2, 1);

  ln1_kernel<<<dim3(MT), dim3(256), 0, stream>>>(x, g1, b1, wwin_f32, wwin_bf, rowmap);

  const dim3 gG(Cdim / 128, MT / 128), gB(256);
  gemm_wmma<EPI_BF16><<<gG, gB, 0, stream>>>(wwin_bf, Bq, nullptr, qb, nullptr,
                                             nullptr, nullptr, MT, Cdim, Cdim);
  gemm_wmma<EPI_BF16><<<gG, gB, 0, stream>>>(wwin_bf, Bk, nullptr, kb, nullptr,
                                             nullptr, nullptr, MT, Cdim, Cdim);
  gemm_wmma<EPI_BF16><<<gG, gB, 0, stream>>>(wwin_bf, Bv, nullptr, vb, nullptr,
                                             nullptr, nullptr, MT, Cdim, Cdim);

  attn_kernel<<<dim3(NWIN * Hn), dim3(128), 0, stream>>>(qb, kb, vb, obuf);

  gemm_wmma<EPI_BIAS_RES_SCAT><<<gG, gB, 0, stream>>>(obuf, Bo, m_f32, nullptr,
                                                      bo, wwin_f32, rowmap,
                                                      MT, Cdim, Cdim);

  ln2_kernel<<<dim3(MT), dim3(256), 0, stream>>>(m_f32, g2, b2, mln);

  gemm_wmma<EPI_GELU_BF16><<<gG, gB, 0, stream>>>(mln, B1, nullptr, h1, bf1,
                                                  nullptr, nullptr, MT, Cdim, Cdim);

  gemm_wmma<EPI_BIAS_RES><<<gG, gB, 0, stream>>>(h1, B2, out, nullptr, bf2,
                                                 m_f32, nullptr, MT, Cdim, Cdim);
}
